// CriterionMiniBatchCrossImagePair_7619271983296
// MI455X (gfx1250) — compile-verified, matrix operand order checked by reading, not run on hardware
//
#include <hip/hip_runtime.h>
#include <hip/hip_bf16.h>

typedef __attribute__((ext_vector_type(16))) _Float16 v16h;
typedef __attribute__((ext_vector_type(8)))  float    v8f;

#define BATCH   4
#define CHAN    256
#define HW      128
#define NPOS    1024        // 32*32 pooled positions
#define INV_TEMP 10.0f      // 1/0.1

// ---------------------------------------------------------------------------
// Kernel 0: zero the scalar output (harness poisons d_out with 0xAA).
// ---------------------------------------------------------------------------
__global__ void zero_out_kernel(float* out) {
    if (threadIdx.x == 0 && blockIdx.x == 0) out[0] = 0.0f;
}

// ---------------------------------------------------------------------------
// Kernel 1: fused 4x4 avg-pool + L2-normalize over C, emit f16 features
//           layout [B, N, C] (C contiguous -> contiguous WMMA K loads).
// One block (256 threads == C) per (b, n) output position.
// ---------------------------------------------------------------------------
__global__ void prep_kernel(const float* __restrict__ in,
                            _Float16* __restrict__ out) {
    __shared__ float red[9];
    const int bn = blockIdx.x;            // 0 .. B*N-1
    const int b  = bn >> 10;
    const int n  = bn & (NPOS - 1);
    const int py = n >> 5;
    const int px = n & 31;
    const int c  = threadIdx.x;           // 0 .. 255

    // 4 rows of 4 contiguous floats (float4 each)
    const float* base = in + (((size_t)(b * CHAN + c) * HW) + py * 4) * HW + px * 4;
    float sum = 0.0f;
#pragma unroll
    for (int dy = 0; dy < 4; ++dy) {
        float4 v = *(const float4*)(base + (size_t)dy * HW);
        sum += v.x + v.y + v.z + v.w;
    }
    const float avg = sum * (1.0f / 16.0f);

    // block-wide sum of squares (wave32 shuffle + LDS across 8 waves)
    float ssq = avg * avg;
#pragma unroll
    for (int o = 16; o > 0; o >>= 1) ssq += __shfl_xor(ssq, o, 32);
    const int wid  = threadIdx.x >> 5;
    const int lane = threadIdx.x & 31;
    if (lane == 0) red[wid] = ssq;
    __syncthreads();
    if (threadIdx.x == 0) {
        float t = 0.0f;
#pragma unroll
        for (int w = 0; w < 8; ++w) t += red[w];
        red[8] = 1.0f / fmaxf(sqrtf(t), 1e-12f);
    }
    __syncthreads();
    const float inv = red[8];

    out[(size_t)bn * CHAN + c] = (_Float16)(avg * inv);
}

// ---------------------------------------------------------------------------
// Kernel 2: fused Gram-tile GEMM (WMMA f16->f32) + row softmax stats + KL.
// grid = B*B*(N/16) blocks, 256 threads (8 waves).
// Dynamic LDS: sims_S[16][1024] + sims_T[16][1024] = 128 KB (fits 320 KB WGP,
// 2 workgroups/WGP). A fragments (16x256 f16) hoisted once per tensor; the
// full B fragment is loaded before the WMMA chain so loads overlap WMMAs.
// ---------------------------------------------------------------------------
__global__ void gram_kl_kernel(const _Float16* __restrict__ fS,
                               const _Float16* __restrict__ fT,
                               float* __restrict__ out) {
    extern __shared__ float sims[];       // [2][16][1024]

    const int blk   = blockIdx.x;
    const int pair  = blk >> 6;           // 0..15
    const int i     = pair >> 2;          // row image
    const int j     = pair & 3;           // col image
    const int n0    = (blk & 63) << 4;    // row-tile base within N

    const int wid   = threadIdx.x >> 5;
    const int lane  = threadIdx.x & 31;
    const int lhalf = lane >> 4;          // 0: lanes 0-15, 1: lanes 16-31
    const int l15   = lane & 15;
    const int kbase = lhalf << 4;         // 16 contiguous halves per lane-half

    // ---------------- Phase 1: 2 x 64 tile-GEMMs, 8 m-tiles per wave --------
#pragma unroll
    for (int tsel = 0; tsel < 2; ++tsel) {
        const _Float16* F = tsel ? fT : fS;

        // Hoist A: row-tile of image i, 8 K-chunks of v16h (64 VGPRs/lane).
        const _Float16* arow = F + ((size_t)i * NPOS + n0 + l15) * CHAN;
        v16h afrag[8];
#pragma unroll
        for (int kk = 0; kk < 8; ++kk)
            afrag[kk] = *(const v16h*)(arow + kk * 32 + kbase);

        float* dstbase = sims + (size_t)tsel * 16 * 1024 + l15;

        for (int mt = wid; mt < 64; mt += 8) {
            const _Float16* brow = F + ((size_t)j * NPOS + (mt << 4) + l15) * CHAN;

            // speculative prefetch of next m-tile's B rows (global_prefetch)
            if (mt + 8 < 64)
                __builtin_prefetch(brow + (size_t)(8 << 4) * CHAN, 0, 1);

            // Load the whole 256x16 B fragment first (16 b128 loads in flight),
            // then chain the 8 WMMAs back-to-back; later loads hide under
            // earlier WMMAs via partial s_wait_loadcnt.
            v16h bfrag[8];
#pragma unroll
            for (int kk = 0; kk < 8; ++kk)
                bfrag[kk] = *(const v16h*)(brow + kk * 32 + kbase);

            v8f acc = {};
#pragma unroll
            for (int kk = 0; kk < 8; ++kk)              // K = 256 in chunks of 32
                acc = __builtin_amdgcn_wmma_f32_16x16x32_f16(
                        false, afrag[kk], false, bfrag[kk], (short)0, acc,
                        false, false);

            // C/D layout: VGPR r, lanes 0-15 -> M=r ; lanes 16-31 -> M=8+r
            float* dst = dstbase + (mt << 4);
#pragma unroll
            for (int r = 0; r < 8; ++r)
                dst[(size_t)(r + (lhalf << 3)) * 1024] = acc[r] * INV_TEMP;
        }
    }
    __syncthreads();

    // ---------------- Phase 2: per-row logsumexp + KL (2 rows per wave) -----
    const float* srowS = sims + (size_t)(wid * 2) * 1024;
    const float* srowT = srowS + 16 * 1024;

    float klsum = 0.0f;
#pragma unroll
    for (int rr = 0; rr < 2; ++rr) {
        const float* rs = srowS + (size_t)rr * 1024;
        const float* rt = srowT + (size_t)rr * 1024;

        float mx_s = -3.4e38f, mx_t = -3.4e38f;
        for (int m = lane; m < 1024; m += 32) {
            mx_s = fmaxf(mx_s, rs[m]);
            mx_t = fmaxf(mx_t, rt[m]);
        }
#pragma unroll
        for (int o = 16; o > 0; o >>= 1) {
            mx_s = fmaxf(mx_s, __shfl_xor(mx_s, o, 32));
            mx_t = fmaxf(mx_t, __shfl_xor(mx_t, o, 32));
        }

        float se_s = 0.0f, se_t = 0.0f;
        for (int m = lane; m < 1024; m += 32) {
            se_s += __expf(rs[m] - mx_s);
            se_t += __expf(rt[m] - mx_t);
        }
#pragma unroll
        for (int o = 16; o > 0; o >>= 1) {
            se_s += __shfl_xor(se_s, o, 32);
            se_t += __shfl_xor(se_t, o, 32);
        }
        const float lse_s = mx_s + __logf(se_s);
        const float lse_t = mx_t + __logf(se_t);

        for (int m = lane; m < 1024; m += 32) {
            const float lpt = rt[m] - lse_t;   // log p_t
            const float lps = rs[m] - lse_s;   // log p_s
            klsum += __expf(lpt) * (lpt - lps);
        }
    }
#pragma unroll
    for (int o = 16; o > 0; o >>= 1) klsum += __shfl_xor(klsum, o, 32);

    // mean over B*B pairs, each KL divided by N -> 1 / (N * B*B)
    if (lane == 0)
        atomicAdd(out, klsum * (1.0f / ((float)NPOS * (float)(BATCH * BATCH))));
}

// ---------------------------------------------------------------------------
extern "C" void kernel_launch(void* const* d_in, const int* in_sizes, int n_in,
                              void* d_out, int out_size, void* d_ws, size_t ws_size,
                              hipStream_t stream) {
    const float* feat_S = (const float*)d_in[0];
    const float* feat_T = (const float*)d_in[1];
    float* outp = (float*)d_out;

    // workspace: f16 features [B,N,C] for S then T (2 MB each)
    _Float16* fSh = (_Float16*)d_ws;
    _Float16* fTh = fSh + (size_t)BATCH * NPOS * CHAN;

    zero_out_kernel<<<1, 32, 0, stream>>>(outp);

    prep_kernel<<<BATCH * NPOS, CHAN, 0, stream>>>(feat_S, fSh);
    prep_kernel<<<BATCH * NPOS, CHAN, 0, stream>>>(feat_T, fTh);

    const size_t lds_bytes = 2u * 16u * 1024u * sizeof(float);   // 128 KB
    gram_kl_kernel<<<BATCH * BATCH * (NPOS / 16), 256, lds_bytes, stream>>>(
        fSh, fTh, outp);
}
